// DeltaNet_22488448762204
// MI455X (gfx1250) — compile-verified, matrix operand order checked by reading, not run on hardware
//
#include <hip/hip_runtime.h>
#include <hip/hip_bf16.h>
#include <math.h>

// DeltaNet-style gated linear attention for MI455X (gfx1250, wave32, WMMA).
// Pipeline: bf16 convert -> WMMA projections (LDS double-buffered weight tiles,
//           async global->LDS) -> RoPE+phi -> chunked WMMA scan
//           (state in f32 accumulators, 64x64 bf16 GEMMs per chunk)
//           -> WMMA out-proj -> residual + RMSNorm.

#define HIDN 1024
#define NH   16
#define HD   64
#define NB   4
#define NT   2048
#define CH   64
#define EPSV 1e-6f

typedef __bf16 bf16;
typedef __attribute__((ext_vector_type(16))) __bf16       v16bf;
typedef __attribute__((ext_vector_type(8)))  float        v8f;
typedef __attribute__((ext_vector_type(8)))  unsigned int v8u;
typedef __attribute__((ext_vector_type(4)))  unsigned int v4u;
typedef __attribute__((__vector_size__(16))) int          i128v;   // v4i

#if __has_builtin(__builtin_amdgcn_global_load_async_to_lds_b128) && \
    __has_builtin(__builtin_amdgcn_s_wait_asynccnt)
#define HAS_ASYNC_LDS 1
#else
#define HAS_ASYNC_LDS 0
#endif

// Copy 16 bytes global -> LDS. Async path uses GLOBAL_LOAD_ASYNC_TO_LDS_B128
// (ASYNCcnt); fallback is a plain load + ds_store (made visible by barriers).
// Builtin params (from clang diagnostic): (v4i AS1*, v4i AS3*, imm, imm).
__device__ __forceinline__ void cp16_to_lds(const bf16* g, bf16* l) {
#if HAS_ASYNC_LDS
  __builtin_amdgcn_global_load_async_to_lds_b128(
      (__attribute__((address_space(1))) i128v*)(g),
      (__attribute__((address_space(3))) i128v*)(l), 0, 0);
#else
  *(v4u*)l = *(const v4u*)g;
#endif
}
__device__ __forceinline__ void async_wait_all() {
#if HAS_ASYNC_LDS
  __builtin_amdgcn_s_wait_asynccnt(0);
#endif
}

__device__ __forceinline__ bf16 f2bf(float x) {
  unsigned u = __builtin_bit_cast(unsigned, x);
  unsigned r = u + 0x7FFFu + ((u >> 16) & 1u);          // round-to-nearest-even
  unsigned short h = (unsigned short)(r >> 16);
  return __builtin_bit_cast(bf16, h);
}
__device__ __forceinline__ float bf2f(bf16 x) {
  unsigned short s = __builtin_bit_cast(unsigned short, x);
  return __builtin_bit_cast(float, ((unsigned)s) << 16);
}

// K-index packing for 16-bit A/B fragments of v_wmma_*_16x16x32 (ISA 7.12.2):
// lane half 0 covers K 0-7 & 16-23, half 1 covers K 8-15 & 24-31, pairs per dword.
__device__ __forceinline__ int kpair(int j, int half) {
  return 8 * half + ((j < 4) ? (2 * j) : (16 + 2 * (j - 4)));
}

// A (or B read as W[n][k]) fragment, matrix row-major [row][k], k contiguous.
__device__ __forceinline__ v16bf ld_frag_rm(const bf16* base, int ld, int row0, int k0) {
  int lane = threadIdx.x & 31;
  int half = lane >> 4;
  const bf16* p = base + (size_t)(row0 + (lane & 15)) * ld + k0;
  v8u u;
#pragma unroll
  for (int j = 0; j < 8; ++j)
    u[j] = *(const unsigned int*)(p + kpair(j, half));   // two packed bf16
  return __builtin_bit_cast(v16bf, u);
}

// Fragment where the WMMA K dimension strides through memory rows:
// element(k, c) = base[k*ld + c]  (e.g. B[k][n] stored row-major in k).
__device__ __forceinline__ v16bf ld_frag_tr(const bf16* base, int ld, int col0, int k0) {
  int lane = threadIdx.x & 31;
  int half = lane >> 4;
  int c = col0 + (lane & 15);
  v16bf a;
#pragma unroll
  for (int j = 0; j < 8; ++j) {
    int k = k0 + kpair(j, half);
    a[2 * j]     = base[(size_t)k * ld + c];
    a[2 * j + 1] = base[(size_t)(k + 1) * ld + c];
  }
  return a;
}

#define WMMA_BF16(A, B, C) \
  __builtin_amdgcn_wmma_f32_16x16x32_bf16(false, (A), false, (B), (short)0, (C), false, false)

// ---------------------------------------------------------------- conversions
__global__ void k_cvt_bf16(const float* __restrict__ src, bf16* __restrict__ dst, int n) {
  for (int i = blockIdx.x * blockDim.x + threadIdx.x; i < n; i += gridDim.x * blockDim.x)
    dst[i] = f2bf(src[i]);
}

// ------------------------------------------------------- C = A @ W^T (+bias)
// A:[M,K] bf16, W:[N,K] bf16, C:[M,N] f32. Block 256 = 8 waves,
// tile 128(M) x 64(N). Weight k-tiles (64x32 bf16 = 4KB) are staged in LDS,
// double-buffered, fetched with async global->LDS; A fragments are register
// double-buffered so global latency hides behind the WMMAs.
__global__ void __launch_bounds__(256) k_gemm_bf16(
    const bf16* __restrict__ A, const bf16* __restrict__ W,
    const float* __restrict__ bias, float* __restrict__ C,
    int M, int N, int K) {
  __shared__ alignas(16) bf16 Bt[2][64 * 32];
  int tid = threadIdx.x;
  int wave = tid >> 5, lane = tid & 31;
  int half = lane >> 4, ln = lane & 15;
  int row0 = blockIdx.y * 128 + wave * 16;
  int col0 = blockIdx.x * 64;

  // cooperative B-tile fetch: 256 threads x 16B = 4KB tile
  int trow = tid >> 2;            // 0..63 (n within tile)
  int tseg = (tid & 3) * 8;       // bf16 offset within 32-wide k row
  const bf16* wsrc = W + (size_t)(col0 + trow) * K + tseg;
  bf16* ldst = &Bt[0][trow * 32 + tseg];

  v8f acc[4];
#pragma unroll
  for (int nt = 0; nt < 4; ++nt)
#pragma unroll
    for (int i = 0; i < 8; ++i) acc[nt][i] = 0.f;

  cp16_to_lds(wsrc, ldst);                    // k-tile 0 -> buffer 0
  async_wait_all();
  __syncthreads();

  v16bf a_cur = ld_frag_rm(A, K, row0, 0);
  for (int k0 = 0; k0 < K; k0 += 32) {
    int buf = (k0 >> 5) & 1;
    if (k0 + 32 < K)                          // prefetch next weight tile
      cp16_to_lds(wsrc + k0 + 32, ldst + (buf ^ 1) * (64 * 32));
    v16bf a_nxt = a_cur;
    if (k0 + 32 < K)                          // prefetch next A fragment
      a_nxt = ld_frag_rm(A, K, row0, k0 + 32);
    const bf16* bt = &Bt[buf][0];
#pragma unroll
    for (int nt = 0; nt < 4; ++nt) {
      v16bf b = ld_frag_rm(bt, 32, 16 * nt, 0);   // B[k][n] = W[n][k] from LDS
      acc[nt] = WMMA_BF16(a_cur, b, acc[nt]);
    }
    a_cur = a_nxt;
    async_wait_all();
    __syncthreads();
  }
#pragma unroll
  for (int nt = 0; nt < 4; ++nt)
#pragma unroll
    for (int i = 0; i < 8; ++i) {
      int r = row0 + i + 8 * half;
      int cc = col0 + 16 * nt + ln;
      float v = acc[nt][i];
      if (bias) v += bias[cc];
      C[(size_t)r * N + cc] = v;
    }
}

// ------------------------------------------------------------- beta = σ(xWbᵀ+bb)
__global__ void k_beta(const float* __restrict__ x, const float* __restrict__ Wb,
                       const float* __restrict__ bbias, float* __restrict__ beta) {
  __shared__ float red[128];
  int row = blockIdx.x;
  int tid = threadIdx.x;
  int h = tid >> 3, sub = tid & 7;
  const float* xr = x + (size_t)row * HIDN;
  const float* wr = Wb + (size_t)h * HIDN;
  float s = 0.f;
  for (int k = sub; k < HIDN; k += 8) s += xr[k] * wr[k];
  red[tid] = s;
  __syncthreads();
  if (sub == 0) {
    float t = 0.f;
#pragma unroll
    for (int j = 0; j < 8; ++j) t += red[h * 8 + j];
    t += bbias[h];
    float sg = 1.f / (1.f + expf(-t));
    sg = fminf(fmaxf(sg, 0.8f), 0.999f);
    beta[(size_t)row * NH + h] = sg;
  }
}

// ------------------------------------------------ RoPE + phi (elu+1) -> bf16
__global__ void k_rope_phi(const float* __restrict__ qf, const float* __restrict__ kf,
                           bf16* __restrict__ pq, bf16* __restrict__ pk) {
  const float LOG1E4 = 9.210340371976184f;  // ln(10000)
  int n = NB * NT * NH * 32;                // pairs (d, d+32)
  for (int p = blockIdx.x * blockDim.x + threadIdx.x; p < n; p += gridDim.x * blockDim.x) {
    int row = p >> 9;                       // b*T + t
    int rem = p & 511;
    int h = rem >> 5, i = rem & 31;
    int pos = row % NT;
    float inv = expf(-(2.f * (float)i / 64.f) * LOG1E4);
    float ang = (float)pos * inv;
    float cs = cosf(ang), sn = sinf(ang);
    size_t lo = (size_t)row * HIDN + h * HD + i;
    size_t hi = lo + 32;
    float ql = qf[lo], qh = qf[hi];
    float rl = ql * cs - qh * sn, rh = qh * cs + ql * sn;
    pq[lo] = f2bf(rl > 0.f ? rl + 1.f : expf(rl));
    pq[hi] = f2bf(rh > 0.f ? rh + 1.f : expf(rh));
    float kl = kf[lo], kh = kf[hi];
    float sl = kl * cs - kh * sn, sh = kh * cs + kl * sn;
    pk[lo] = f2bf(sl > 0.f ? sl + 1.f : expf(sl));
    pk[hi] = f2bf(sh > 0.f ? sh + 1.f : expf(sh));
  }
}

// ------------------------------------------------- chunked gated linear scan
// One block per (b,h); 4 waves. S (64x64 f32) lives in wave accumulators.
__global__ void __launch_bounds__(128) k_scan(
    const bf16* __restrict__ pq, const bf16* __restrict__ pk,
    const bf16* __restrict__ vb, const float* __restrict__ beta,
    float* __restrict__ attn) {
  __shared__ alignas(16) bf16 qs[CH * HD], ks[CH * HD], vs[CH * HD];
  __shared__ alignas(16) bf16 Sb[HD * HD], Pb[CH * CH];
  __shared__ float cdec[CH], cinv[CH], zvec[HD], denI[CH], denT[CH], bch[CH];

  int b = blockIdx.x / NH, h = blockIdx.x % NH;
  int tid = threadIdx.x;
  int wave = tid >> 5, lane = tid & 31, half = lane >> 4, ln = lane & 15;

  v8f Sreg[4];
#pragma unroll
  for (int nt = 0; nt < 4; ++nt)
#pragma unroll
    for (int i = 0; i < 8; ++i) Sreg[nt][i] = 0.f;
  if (tid < HD) zvec[tid] = 0.f;
  __syncthreads();

  for (int t0 = 0; t0 < NT; t0 += CH) {
    // async-stage raw V chunk while betas / decays are prepared
    {
      const bf16* vbase = vb + ((size_t)b * NT + t0) * HIDN + h * HD;
      for (int c = tid; c < (CH * HD) / 8; c += 128) {   // 512 x 16B
        int t = c >> 3, seg = (c & 7) * 8;
        cp16_to_lds(vbase + (size_t)t * HIDN + seg, &vs[t * HD + seg]);
      }
    }
    if (tid < CH) bch[tid] = beta[((size_t)b * NT + t0 + tid) * NH + h];
    __syncthreads();
    if (tid == 0) {
      float run = 1.f;
      for (int t = 0; t < CH; ++t) { run *= bch[t]; cdec[t] = run; cinv[t] = 1.f / run; }
    }
    __syncthreads();

    // Stage chunk: q~ = c_t*phi_q, k~ = phi_k/c_t; dump S0 as bf16.
    for (int e = tid; e < CH * HD; e += 128) {
      int t = e >> 6, d = e & 63;
      size_t g = ((size_t)b * NT + t0 + t) * HIDN + h * HD + d;
      qs[e] = f2bf(bf2f(pq[g]) * cdec[t]);
      ks[e] = f2bf(bf2f(pk[g]) * cinv[t]);
    }
#pragma unroll
    for (int nt = 0; nt < 4; ++nt)
#pragma unroll
      for (int i = 0; i < 8; ++i)
        Sb[(16 * wave + i + 8 * half) * HD + 16 * nt + ln] = f2bf(Sreg[nt][i]);
    async_wait_all();
    __syncthreads();

    // Inter denominator: c_t * phi_q . z0
    if (tid < CH) {
      float s = 0.f;
      for (int d = 0; d < HD; ++d) s += bf2f(qs[tid * HD + d]) * zvec[d];
      denT[tid] = s;
    }

    // On = q~ @ S0 ; P = q~ @ k~^T (shared A fragment).
    v8f On[4], Pa[4];
#pragma unroll
    for (int nt = 0; nt < 4; ++nt)
#pragma unroll
      for (int i = 0; i < 8; ++i) { On[nt][i] = 0.f; Pa[nt][i] = 0.f; }
#pragma unroll
    for (int k0 = 0; k0 < HD; k0 += 32) {
      v16bf aq = ld_frag_rm(qs, HD, 16 * wave, k0);
#pragma unroll
      for (int nt = 0; nt < 4; ++nt) {
        v16bf bs = ld_frag_tr(Sb, HD, 16 * nt, k0);   // B[d][e] = Sb[d][e]
        On[nt] = WMMA_BF16(aq, bs, On[nt]);
        v16bf bk = ld_frag_rm(ks, HD, 16 * nt, k0);   // B[d][s] = ks[s][d]
        Pa[nt] = WMMA_BF16(aq, bk, Pa[nt]);
      }
    }

    // Causal mask, intra denominator (row sums), store P as bf16.
    float rsum[8];
#pragma unroll
    for (int i = 0; i < 8; ++i) rsum[i] = 0.f;
#pragma unroll
    for (int nt = 0; nt < 4; ++nt)
#pragma unroll
      for (int i = 0; i < 8; ++i) {
        int t = 16 * wave + i + 8 * half;
        int s = 16 * nt + ln;
        float v = (s <= t) ? Pa[nt][i] : 0.f;
        rsum[i] += v;
        Pb[t * CH + s] = f2bf(v);
      }
#pragma unroll
    for (int i = 0; i < 8; ++i) {
      float r = rsum[i];
      r += __shfl_xor(r, 1);
      r += __shfl_xor(r, 2);
      r += __shfl_xor(r, 4);
      r += __shfl_xor(r, 8);   // reduce over 16 lanes within half
      if (ln == 0) denI[16 * wave + i + 8 * half] = r;
    }
    __syncthreads();

    // On += P @ V
#pragma unroll
    for (int k0 = 0; k0 < CH; k0 += 32) {
      v16bf ap = ld_frag_rm(Pb, CH, 16 * wave, k0);
#pragma unroll
      for (int nt = 0; nt < 4; ++nt) {
        v16bf bv = ld_frag_tr(vs, HD, 16 * nt, k0);   // B[s][e] = vs[s][e]
        On[nt] = WMMA_BF16(ap, bv, On[nt]);
      }
    }

    // Write out = On / (den_inter + den_intra + eps)
#pragma unroll
    for (int nt = 0; nt < 4; ++nt)
#pragma unroll
      for (int i = 0; i < 8; ++i) {
        int t = 16 * wave + i + 8 * half;
        int cc = 16 * nt + ln;
        float den = denT[t] + denI[t] + EPSV;
        attn[((size_t)b * NT + t0 + t) * HIDN + h * HD + cc] = On[nt][i] / den;
      }

    // State: S = cL * (S + k~^T @ V) ; z = cL * (z + sum_t k~_t)
#pragma unroll
    for (int k0 = 0; k0 < CH; k0 += 32) {
      v16bf ak = ld_frag_tr(ks, HD, 16 * wave, k0);   // A[d][s] = ks[s][d]
#pragma unroll
      for (int nt = 0; nt < 4; ++nt) {
        v16bf bv = ld_frag_tr(vs, HD, 16 * nt, k0);
        Sreg[nt] = WMMA_BF16(ak, bv, Sreg[nt]);
      }
    }
    float cL = cdec[CH - 1];
#pragma unroll
    for (int nt = 0; nt < 4; ++nt)
#pragma unroll
      for (int i = 0; i < 8; ++i) Sreg[nt][i] *= cL;
    if (tid < HD) {
      float s = 0.f;
      for (int t = 0; t < CH; ++t) s += bf2f(ks[t * HD + tid]);
      zvec[tid] = cL * (zvec[tid] + s);
    }
    __syncthreads();
  }
}

// ----------------------------------------------- residual + RMSNorm -> d_out
__global__ void k_rmsnorm(const float* __restrict__ x, const float* __restrict__ y,
                          const float* __restrict__ scale, float* __restrict__ out) {
  __shared__ float red[256];
  int row = blockIdx.x, tid = threadIdx.x;
  const float* xr = x + (size_t)row * HIDN;
  const float* yr = y + (size_t)row * HIDN;
  float hv[4];
  float ss = 0.f;
#pragma unroll
  for (int j = 0; j < 4; ++j) {
    int c = tid + 256 * j;
    float hh = xr[c] + yr[c];
    hv[j] = hh;
    ss += hh * hh;
  }
  red[tid] = ss;
  __syncthreads();
  for (int off = 128; off > 0; off >>= 1) {
    if (tid < off) red[tid] += red[tid + off];
    __syncthreads();
  }
  float rinv = rsqrtf(red[0] / (float)HIDN + EPSV);
#pragma unroll
  for (int j = 0; j < 4; ++j) {
    int c = tid + 256 * j;
    out[(size_t)row * HIDN + c] = hv[j] * rinv * scale[c];
  }
}

// ---------------------------------------------------------------------- host
extern "C" void kernel_launch(void* const* d_in, const int* in_sizes, int n_in,
                              void* d_out, int out_size, void* d_ws, size_t ws_size,
                              hipStream_t stream) {
  (void)in_sizes; (void)n_in; (void)out_size; (void)ws_size;
  const float* x     = (const float*)d_in[0];
  const float* Wq    = (const float*)d_in[1];
  const float* Wk    = (const float*)d_in[2];
  const float* Wv    = (const float*)d_in[3];
  const float* Wb    = (const float*)d_in[4];
  const float* bb    = (const float*)d_in[5];
  const float* Wo    = (const float*)d_in[6];
  const float* bo    = (const float*)d_in[7];
  const float* scale = (const float*)d_in[8];
  float* out = (float*)d_out;

  char* ws = (char*)d_ws;
  size_t off = 0;
  auto take = [&](size_t bytes) -> void* {
    void* p = (void*)(ws + off);
    off += (bytes + 255) & ~(size_t)255;
    return p;
  };
  const size_t nX = (size_t)NB * NT * HIDN, nW = (size_t)HIDN * HIDN;
  bf16*  xb    = (bf16*)take(nX * 2);        // x (later: attn) in bf16
  bf16*  Wqb   = (bf16*)take(nW * 2);
  bf16*  Wkb   = (bf16*)take(nW * 2);
  bf16*  Wvb   = (bf16*)take(nW * 2);
  bf16*  Wob   = (bf16*)take(nW * 2);
  bf16*  pq    = (bf16*)take(nX * 2);        // phi(rope(q))
  bf16*  pk    = (bf16*)take(nX * 2);        // phi(rope(k))
  bf16*  vbuf  = (bf16*)take(nX * 2);        // v in bf16
  float* betab = (float*)take((size_t)NB * NT * NH * 4);
  float* buf0  = (float*)take(nX * 4);       // q / v / attn scratch
  float* buf1  = (float*)take(nX * 4);       // k / y scratch

  k_cvt_bf16<<<2048, 256, 0, stream>>>(x, xb, (int)nX);
  k_cvt_bf16<<<512, 256, 0, stream>>>(Wq, Wqb, (int)nW);
  k_cvt_bf16<<<512, 256, 0, stream>>>(Wk, Wkb, (int)nW);
  k_cvt_bf16<<<512, 256, 0, stream>>>(Wv, Wvb, (int)nW);
  k_cvt_bf16<<<512, 256, 0, stream>>>(Wo, Wob, (int)nW);

  dim3 ggrid(HIDN / 64, (NB * NT) / 128);    // (16, 64)
  k_gemm_bf16<<<ggrid, 256, 0, stream>>>(xb, Wqb, nullptr, buf0, NB * NT, HIDN, HIDN);
  k_gemm_bf16<<<ggrid, 256, 0, stream>>>(xb, Wkb, nullptr, buf1, NB * NT, HIDN, HIDN);
  k_beta<<<NB * NT, 128, 0, stream>>>(x, Wb, bb, betab);
  k_rope_phi<<<2048, 256, 0, stream>>>(buf0, buf1, pq, pk);
  k_gemm_bf16<<<ggrid, 256, 0, stream>>>(xb, Wvb, nullptr, buf0, NB * NT, HIDN, HIDN);
  k_cvt_bf16<<<2048, 256, 0, stream>>>(buf0, vbuf, (int)nX);

  k_scan<<<NB * NH, 128, 0, stream>>>(pq, pk, vbuf, betab, buf0);  // attn (f32)

  k_cvt_bf16<<<2048, 256, 0, stream>>>(buf0, xb, (int)nX);          // attn -> bf16
  k_gemm_bf16<<<ggrid, 256, 0, stream>>>(xb, Wob, bo, buf1, NB * NT, HIDN, HIDN);
  k_rmsnorm<<<NB * NT, 256, 0, stream>>>(x, buf1, scale, out);
}
